// KNRM_class_80625126081187
// MI455X (gfx1250) — matrix operand order repaired and checked by reading, not compiled, not run on hardware
//
#include <hip/hip_runtime.h>
#include <hip/hip_bf16.h>
#include <math.h>

typedef __attribute__((ext_vector_type(16))) _Float16 v16h;
typedef __attribute__((ext_vector_type(8)))  float    v8f;

#define EDIM 300
#define NKB  10        // ceil(300/32) -> padded K = 320
#define QL   16
#define DL   1024
#define NK   11

// ---------------------------------------------------------------------------
// Kernel 0: reciprocal L2 norms of query embeddings.  One wave per (b,q).
// ---------------------------------------------------------------------------
__global__ __launch_bounds__(32) void knrm_qnorm(const int* __restrict__ qtoks,
                                                 const float* __restrict__ emb,
                                                 float* __restrict__ qrnorm) {
  int bq   = blockIdx.x;        // b*16 + q
  int lane = threadIdx.x;
  int tok  = qtoks[bq];
  const float4* row = (const float4*)(emb + (size_t)tok * EDIM);  // 75 float4s
  float s = 0.f;
  for (int f = lane; f < EDIM / 4; f += 32) {
    float4 v = row[f];
    s += v.x * v.x + v.y * v.y + v.z * v.z + v.w * v.w;
  }
  s += __shfl_xor(s, 16); s += __shfl_xor(s, 8); s += __shfl_xor(s, 4);
  s += __shfl_xor(s, 2);  s += __shfl_xor(s, 1);
  if (lane == 0) qrnorm[bq] = 1.0f / (sqrtf(s) + 1e-9f);
}

// ---------------------------------------------------------------------------
// Kernel 1: build f16 A-fragments (16x32 MxK WMMA layout) for the queries.
// One wave per (b, kb).  Lane L holds row m = L%16; half h maps to
// K = (L<16?0:8) + (h<8 ? h : h+8)   (ISA 7.12.2, 16-bit A 16x32).
// ---------------------------------------------------------------------------
__global__ __launch_bounds__(32) void knrm_qa(const int* __restrict__ qtoks,
                                              const float* __restrict__ emb,
                                              const float* __restrict__ qrnorm,
                                              _Float16* __restrict__ qA) {
  int blk = blockIdx.x;         // b*NKB + kb
  int b   = blk / NKB;
  int kb  = blk % NKB;
  int L   = threadIdx.x;
  int m   = L & 15;
  int tok = qtoks[b * QL + m];
  float rn = qrnorm[b * QL + m];
  const float* row = emb + (size_t)tok * EDIM;
  int kbase = (L < 16) ? 0 : 8;
  v16h frag;
#pragma unroll
  for (int h = 0; h < 16; ++h) {
    int K = kbase + (h < 8 ? h : h + 8);
    int e = kb * 32 + K;
    float v = (e < EDIM) ? row[e] * rn : 0.f;
    frag[h] = (_Float16)v;
  }
  *(v16h*)(qA + ((size_t)blk * 32 + L) * 16) = frag;   // 32B per lane
}

// ---------------------------------------------------------------------------
// Kernel 2: main.  One block per batch b; 8 waves, each owns 8 doc tiles.
// Per tile: gather+normalize 16 doc rows -> f16 B fragments (32x16 KxN
// layout: lane n holds column n, halves = K 8*hi+0..7, then 16+8*hi+0..7), 10 chained
// v_wmma_f32_16x16x32_f16 over E, RBF bank accumulated per-lane in
// registers across the 8 tiles, then ONE butterfly reduction + LDS
// accumulation per wave at the end (8x fewer ds ops than per-tile reduce).
// ---------------------------------------------------------------------------
__global__ __launch_bounds__(256) void knrm_main(const int* __restrict__ dtoks,
                                                 const float* __restrict__ emb,
                                                 const _Float16* __restrict__ qA,
                                                 const float* __restrict__ mus,
                                                 const float* __restrict__ sigmas,
                                                 const float* __restrict__ fc_w,
                                                 const float* __restrict__ fc_b,
                                                 float* __restrict__ out) {
  __shared__ float s_res[NK][QL];
  __shared__ float s_ssum[QL];

  int b    = blockIdx.x;
  int tid  = threadIdx.x;
  int wave = tid >> 5;
  int L    = tid & 31;
  int n    = L & 15;
  bool hi  = (L >= 16);

  if (tid < NK * QL) ((float*)s_res)[tid] = 0.f;
  if (tid < QL)      s_ssum[tid] = 0.f;
  __syncthreads();

  float mu[NK], nis[NK];
#pragma unroll
  for (int k = 0; k < NK; ++k) {
    mu[k] = mus[k];
    float sg = sigmas[k];
    nis[k] = -0.5f / (sg * sg);
  }

  // A fragments for this batch (reused across all 64 tiles)
  v16h a[NKB];
#pragma unroll
  for (int kb = 0; kb < NKB; ++kb)
    a[kb] = *(const v16h*)(qA + ((size_t)(b * NKB + kb) * 32 + L) * 16);

  // Per-lane accumulators across this wave's 8 tiles:
  // acc[r][k] = sum over processed docs (this lane's columns) of RBF kernel k
  // ss[r]     = sum of sims (for the log mask)
  float acc[8][NK];
  float ss[8];
#pragma unroll
  for (int r = 0; r < 8; ++r) {
    ss[r] = 0.f;
#pragma unroll
    for (int k = 0; k < NK; ++k) acc[r][k] = 0.f;
  }

  for (int ti = 0; ti < 8; ++ti) {
    int t   = wave * 8 + ti;               // tile 0..63
    int d   = t * 16 + n;                  // doc column owned by this lane
    int tok = dtoks[b * DL + d];
    const float* row = emb + (size_t)tok * EDIM;

    // L2 norm of doc row: lanes L and L+16 split the 75 float4 blocks
    float s = 0.f;
    for (int f = hi ? 1 : 0; f < EDIM / 4; f += 2) {
      float4 v = ((const float4*)row)[f];
      s += v.x * v.x + v.y * v.y + v.z * v.z + v.w * v.w;
    }
    s += __shfl_xor(s, 16);
    float rn = 1.0f / (sqrtf(s) + 1e-9f);

    // GEMM over E: D = A(16x32) * B(32x16) accumulated in f32
    v8f c = {};
#pragma unroll
    for (int kb = 0; kb < NKB; ++kb) {
      int ebase = kb * 32 + (hi ? 8 : 0);  // K runs ebase..+7 and ebase+16..+23 per lane
      v16h bf;
#pragma unroll
      for (int j = 0; j < 4; ++j) {
        int e0 = ebase + (j < 2 ? j * 4 : 16 + (j - 2) * 4);
        float4 v;
        if (e0 + 3 < EDIM) {
          v = *(const float4*)(row + e0);  // 16B aligned: rows are 75 float4s
        } else {
          v.x = (e0 + 0 < EDIM) ? row[e0 + 0] : 0.f;
          v.y = (e0 + 1 < EDIM) ? row[e0 + 1] : 0.f;
          v.z = (e0 + 2 < EDIM) ? row[e0 + 2] : 0.f;
          v.w = (e0 + 3 < EDIM) ? row[e0 + 3] : 0.f;
        }
        bf[j * 4 + 0] = (_Float16)(v.x * rn);
        bf[j * 4 + 1] = (_Float16)(v.y * rn);
        bf[j * 4 + 2] = (_Float16)(v.z * rn);
        bf[j * 4 + 3] = (_Float16)(v.w * rn);
      }
      c = __builtin_amdgcn_wmma_f32_16x16x32_f16(
          /*neg_a=*/false, a[kb], /*neg_b=*/false, bf,
          /*c_mod=*/(short)0, c, /*reuse_a=*/false, /*reuse_b=*/false);
    }

    // RBF kernel bank, accumulated per-lane (no cross-lane traffic here).
    // C layout: lane holds column n, rows m = r + (hi ? 8 : 0).
#pragma unroll
    for (int r = 0; r < 8; ++r) {
      float sim = c[r];
      ss[r] += sim;
#pragma unroll
      for (int k = 0; k < NK; ++k) {
        float df = sim - mu[k];
        acc[r][k] += __expf(df * df * nis[k]);
      }
    }
  }

  // One reduction per wave: butterfly over the 16 columns, then LDS atomics.
#pragma unroll
  for (int r = 0; r < 8; ++r) {
    int m = r + (hi ? 8 : 0);
    float srow = ss[r];
    srow += __shfl_xor(srow, 1); srow += __shfl_xor(srow, 2);
    srow += __shfl_xor(srow, 4); srow += __shfl_xor(srow, 8);
    if (n == 0) atomicAdd(&s_ssum[m], srow);
#pragma unroll
    for (int k = 0; k < NK; ++k) {
      float e = acc[r][k];
      e += __shfl_xor(e, 1); e += __shfl_xor(e, 2);
      e += __shfl_xor(e, 4); e += __shfl_xor(e, 8);
      if (n == 0) atomicAdd(&s_res[k][m], e);
    }
  }
  __syncthreads();

  // Epilogue: masked log, sum over Q, 11->1 FC.  Trivial work, thread 0.
  if (tid == 0) {
    float score = fc_b[0];
#pragma unroll
    for (int k = 0; k < NK; ++k) {
      float a2 = 0.f;
      for (int q = 0; q < QL; ++q)
        if (s_ssum[q] != 0.f) a2 += __logf(s_res[k][q] + 1e-6f);
      score += fc_w[k] * a2;
    }
    out[b] = score;
  }
}

// ---------------------------------------------------------------------------
extern "C" void kernel_launch(void* const* d_in, const int* in_sizes, int n_in,
                              void* d_out, int out_size, void* d_ws, size_t ws_size,
                              hipStream_t stream) {
  const int*   dtoks = (const int*)d_in[0];    // [B,1024]
  const int*   qtoks = (const int*)d_in[1];    // [B,16]
  // d_in[2] = query_idf : unused by the reference
  const float* emb   = (const float*)d_in[3];  // [100000,300]
  const float* mus   = (const float*)d_in[4];  // [11]
  const float* sig   = (const float*)d_in[5];  // [11]
  const float* fcw   = (const float*)d_in[6];  // [1,11]
  const float* fcb   = (const float*)d_in[7];  // [1]
  float* out = (float*)d_out;

  int B = in_sizes[1] / QL;                    // 128

  float*    qrnorm = (float*)d_ws;                           // B*16 floats (8 KB)
  _Float16* qA     = (_Float16*)((char*)d_ws + 8192);        // B*10*32*16 halves (~1.25 MB)

  knrm_qnorm<<<B * QL,  32, 0, stream>>>(qtoks, emb, qrnorm);
  knrm_qa   <<<B * NKB, 32, 0, stream>>>(qtoks, emb, qrnorm, qA);
  knrm_main <<<B,      256, 0, stream>>>(dtoks, emb, qA, mus, sig, fcw, fcb, out);
}
